// GIC_72310069395602
// MI455X (gfx1250) — matrix operand order repaired and checked by reading, not compiled, not run on hardware
//
#include <hip/hip_runtime.h>
#include <cstdint>
#include <cstddef>

// ---------------- problem constants ----------------
#define N_NODES 100000
#define HDIM    512
#define KCLUS   128
#define BETA_T  100.0f

// padded row counts: NPAD covers the 256-row dist blocks, NPAD2 the split-K chunks
#define NPAD    100096          // 391 * 256
#define NCHUNK  2048
#define NCHUNKS 49
#define NPAD2   (NCHUNK * NCHUNKS)   // 100352
#define HBLOCKS 4

typedef __bf16 bf16_t;
typedef __attribute__((ext_vector_type(16))) __bf16 v16bf;
typedef __attribute__((ext_vector_type(8)))  __bf16 v8bf;
typedef __attribute__((ext_vector_type(4)))  __bf16 v4bf;
typedef __attribute__((ext_vector_type(8)))  float  v8f;

// ---------------- helpers ----------------
__device__ __forceinline__ float red16_add(float v) {
    v += __shfl_xor(v, 1, 32); v += __shfl_xor(v, 2, 32);
    v += __shfl_xor(v, 4, 32); v += __shfl_xor(v, 8, 32);
    return v;
}
__device__ __forceinline__ float red16_max(float v) {
    v = fmaxf(v, __shfl_xor(v, 1, 32)); v = fmaxf(v, __shfl_xor(v, 2, 32));
    v = fmaxf(v, __shfl_xor(v, 4, 32)); v = fmaxf(v, __shfl_xor(v, 8, 32));
    return v;
}
__device__ __forceinline__ float red32_add(float v) {
    v = red16_add(v); v += __shfl_xor(v, 16, 32);
    return v;
}
__device__ __forceinline__ float softplusf(float x) {
    return fmaxf(x, 0.0f) + __logf(1.0f + __expf(-fabsf(x)));
}
__device__ __forceinline__ float sigmoidf_(float x) {
    return 1.0f / (1.0f + __expf(-x));
}
__device__ __forceinline__ v16bf pack16(v8bf lo, v8bf hi) {
    v16bf a;
#pragma unroll
    for (int e = 0; e < 8; ++e) { a[e] = lo[e]; a[e + 8] = hi[e]; }
    return a;
}
__device__ __forceinline__ v16bf wmma_bf16(v16bf A, v16bf B, v8f& acc) {
    acc = __builtin_amdgcn_wmma_f32_16x16x32_bf16(false, A, false, B, (short)0, acc,
                                                  false, false);
    return A;
}

// ---------------- phase 0: normalize rows of `positive` to bf16 ----------------
__global__ void __launch_bounds__(256)
k_normalize(const float* __restrict__ pos, bf16_t* __restrict__ dn)
{
    const long row = (long)((blockIdx.x * 256 + threadIdx.x) >> 5);
    const int lane = threadIdx.x & 31;
    if (row >= N_NODES) return;
    const float* src = pos + row * (long)HDIM;
    float4 v[4];
    float ss = 0.0f;
#pragma unroll
    for (int c = 0; c < 4; ++c) {
        v[c] = ((const float4*)src)[lane + 32 * c];
        ss += v[c].x * v[c].x + v[c].y * v[c].y + v[c].z * v[c].z + v[c].w * v[c].w;
    }
    ss = red32_add(ss);
    const float s = 1.0f / (sqrtf(ss) + 1e-8f);
    bf16_t* orow = dn + row * (long)HDIM;
#pragma unroll
    for (int c = 0; c < 4; ++c) {
        v4bf b;
        b[0] = (bf16_t)(v[c].x * s); b[1] = (bf16_t)(v[c].y * s);
        b[2] = (bf16_t)(v[c].z * s); b[3] = (bf16_t)(v[c].w * s);
        *(v4bf*)(orow + 4 * (lane + 32 * c)) = b;
    }
}

// ---------------- column sums of `positive` (graph summary) ----------------
__global__ void __launch_bounds__(256)
k_colsum(const float* __restrict__ pos, float* __restrict__ colsum)
{
    const int t = threadIdx.x;
    const long r0 = (long)blockIdx.x * 256;
    const long rend = (r0 + 256 < N_NODES) ? r0 + 256 : N_NODES;
    float a0 = 0.0f, a1 = 0.0f;
    for (long i = r0; i < rend; ++i) {
        const float* row = pos + i * (long)HDIM;
        a0 += row[t];
        a1 += row[t + 256];
    }
    atomicAdd(colsum + t, a0);
    atomicAdd(colsum + t + 256, a1);
}

// ---------------- Ws = W @ sigmoid(colsum / N) ----------------
__global__ void __launch_bounds__(256)
k_ws(const float* __restrict__ W, const float* __restrict__ colsum, float* __restrict__ Ws)
{
    const int row = (blockIdx.x * 256 + threadIdx.x) >> 5;
    const int lane = threadIdx.x & 31;
    if (row >= HDIM) return;
    const float invn = 1.0f / (float)N_NODES;
    float acc = 0.0f;
#pragma unroll
    for (int c = 0; c < 4; ++c) {
        float4 wv = ((const float4*)(W + (size_t)row * HDIM))[lane + 32 * c];
        float4 cs = ((const float4*)colsum)[lane + 32 * c];
        acc += wv.x * sigmoidf_(cs.x * invn) + wv.y * sigmoidf_(cs.y * invn)
             + wv.z * sigmoidf_(cs.z * invn) + wv.w * sigmoidf_(cs.w * invn);
    }
    acc = red32_add(acc);
    if (lane == 0) Ws[row] = acc;
}

// ------- per-iteration: mu_n = normalize((mu_acc/cr) or init); zero mu_acc, cr -------
__global__ void __launch_bounds__(256)
k_mu_prep(const float* __restrict__ initmu, float* __restrict__ mu_acc,
          float* __restrict__ cr, bf16_t* __restrict__ mun, int use_init)
{
    const int kk = (blockIdx.x * 256 + threadIdx.x) >> 5;
    const int lane = threadIdx.x & 31;
    if (kk >= KCLUS) return;
    const float* src = use_init ? (initmu + (size_t)kk * HDIM) : (mu_acc + (size_t)kk * HDIM);
    const float invc = use_init ? 1.0f : (1.0f / cr[kk]);
    float4 v[4];
    float ss = 0.0f;
#pragma unroll
    for (int c = 0; c < 4; ++c) {
        v[c] = ((const float4*)src)[lane + 32 * c];
        v[c].x *= invc; v[c].y *= invc; v[c].z *= invc; v[c].w *= invc;
        ss += v[c].x * v[c].x + v[c].y * v[c].y + v[c].z * v[c].z + v[c].w * v[c].w;
    }
    ss = red32_add(ss);
    const float s = 1.0f / sqrtf(ss);           // reference: mu / ||mu|| (no eps)
    bf16_t* orow = mun + (size_t)kk * HDIM;
#pragma unroll
    for (int c = 0; c < 4; ++c) {
        v4bf b;
        b[0] = (bf16_t)(v[c].x * s); b[1] = (bf16_t)(v[c].y * s);
        b[2] = (bf16_t)(v[c].z * s); b[3] = (bf16_t)(v[c].w * s);
        *(v4bf*)(orow + 4 * (lane + 32 * c)) = b;
    }
    const float4 z = make_float4(0.f, 0.f, 0.f, 0.f);
#pragma unroll
    for (int c = 0; c < 4; ++c)
        ((float4*)(mu_acc + (size_t)kk * HDIM))[lane + 32 * c] = z;
    if (lane == 0) cr[kk] = 0.0f;
}

// ------- dist = dn @ mu_nᵀ ; r = softmax(beta*dist) ; cluster_r ; write rᵀ -------
// Each wave handles TWO 16-row A tiles (32 nodes) sharing every B fragment:
// halves B loads / waitcnt stalls per WMMA and doubles per-wave WMMA ILP.
__global__ void __launch_bounds__(256)
k_dist_softmax(const bf16_t* __restrict__ dn, const bf16_t* __restrict__ mun,
               bf16_t* __restrict__ rt, bf16_t* __restrict__ rrm,
               float* __restrict__ cr, int write_rrm)
{
    __shared__ float lds_cr[KCLUS];
    const int t = threadIdx.x;
    if (t < KCLUS) lds_cr[t] = 0.0f;
    __syncthreads();

    const int w = t >> 5, lane = t & 31;
    const int half = lane >> 4, nn = lane & 15;
    const long iw = (long)blockIdx.x * 256 + (long)w * 32;

    v8f acc[2][8];
#pragma unroll
    for (int rr = 0; rr < 2; ++rr)
#pragma unroll
        for (int c = 0; c < 8; ++c)
#pragma unroll
            for (int j = 0; j < 8; ++j) acc[rr][c][j] = 0.0f;

    const bf16_t* arow0 = dn + (iw + nn) * (long)HDIM;
    const bf16_t* arow1 = dn + (iw + 16 + nn) * (long)HDIM;
#pragma unroll 2
    for (int kt = 0; kt < 16; ++kt) {
        const int h0 = kt * 32;
        v16bf A0 = *(const v16bf*)(arow0 + h0 + 16 * half);
        v16bf A1 = *(const v16bf*)(arow1 + h0 + 16 * half);
#pragma unroll
        for (int c = 0; c < 8; ++c) {
            v16bf B = *(const v16bf*)(mun + (c * 16 + nn) * HDIM + h0 + 16 * half);
            acc[0][c] = __builtin_amdgcn_wmma_f32_16x16x32_bf16(false, A0, false, B,
                                                                (short)0, acc[0][c],
                                                                false, false);
            acc[1][c] = __builtin_amdgcn_wmma_f32_16x16x32_bf16(false, A1, false, B,
                                                                (short)0, acc[1][c],
                                                                false, false);
        }
    }

#pragma unroll
    for (int rr = 0; rr < 2; ++rr) {
        const long ib = iw + 16 * rr;

        // softmax over the 128 clusters of each row
#pragma unroll
        for (int j = 0; j < 8; ++j) {
            float m = acc[rr][0][j];
#pragma unroll
            for (int c = 1; c < 8; ++c) m = fmaxf(m, acc[rr][c][j]);
            m = red16_max(m);
            float s = 0.0f;
#pragma unroll
            for (int c = 0; c < 8; ++c) {
                float e = __expf(BETA_T * (acc[rr][c][j] - m));
                acc[rr][c][j] = e; s += e;
            }
            s = red16_add(s);
            const long node = ib + j + 8 * half;
            const float scale = (node < N_NODES) ? (1.0f / s) : 0.0f;
#pragma unroll
            for (int c = 0; c < 8; ++c) acc[rr][c][j] *= scale;
        }

        // cluster_r += column sums of r
#pragma unroll
        for (int c = 0; c < 8; ++c) {
            float s = 0.0f;
#pragma unroll
            for (int j = 0; j < 8; ++j) s += acc[rr][c][j];
            s += __shfl_xor(s, 16, 32);
            if (half == 0) atomicAdd(&lds_cr[c * 16 + nn], s);
        }

        // store r transposed [KCLUS x NPAD2] (contiguous 8-element stores per lane)
#pragma unroll
        for (int c = 0; c < 8; ++c) {
            v8bf v;
#pragma unroll
            for (int j = 0; j < 8; ++j) v[j] = (bf16_t)acc[rr][c][j];
            *(v8bf*)(rt + (size_t)(c * 16 + nn) * NPAD2 + ib + 8 * half) = v;
        }

        if (write_rrm) {     // last iteration: also row-major r for the epilogue GEMM
#pragma unroll
            for (int j = 0; j < 8; ++j) {
                const long node = ib + j + 8 * half;
                if (node < N_NODES) {
#pragma unroll
                    for (int c = 0; c < 8; ++c)
                        rrm[node * (long)KCLUS + c * 16 + nn] = (bf16_t)acc[rr][c][j];
                }
            }
        }
    }

    __syncthreads();
    if (t < KCLUS) atomicAdd(&cr[t], lds_cr[t]);
}

// ------- mu_acc += rᵀ @ dn  (split-K, data tile transposed through LDS) -------
__global__ void __launch_bounds__(256)
k_mu_accum(const bf16_t* __restrict__ rt, const bf16_t* __restrict__ dn,
           float* __restrict__ mu_acc)
{
    __shared__ bf16_t T[KCLUS * 64];   // [h_local 0..127][kk 0..63], 16 KB
    const int t = threadIdx.x, w = t >> 5, lane = t & 31;
    const int half = lane >> 4, nn = lane & 15;
    const long i0 = (long)blockIdx.x * NCHUNK;
    const int h0 = blockIdx.y * 128;

    v8f acc[8];
#pragma unroll
    for (int c = 0; c < 8; ++c)
#pragma unroll
        for (int j = 0; j < 8; ++j) acc[c][j] = 0.0f;

    const bf16_t* arow = rt + (size_t)(16 * w + nn) * NPAD2;

    for (int kb = 0; kb < NCHUNK; kb += 64) {
        __syncthreads();
        {   // cooperative transpose of a 64x128 data tile into LDS
            const int rr = t >> 2, cg = t & 3;
            const bf16_t* src = dn + (i0 + kb + rr) * (long)HDIM + h0 + cg * 32;
            if (kb + 64 < NCHUNK)
                __builtin_prefetch(dn + (i0 + kb + 64 + rr) * (long)HDIM + h0 + cg * 32, 0, 1);
#pragma unroll
            for (int e = 0; e < 32; ++e) T[(cg * 32 + e) * 64 + rr] = src[e];
        }
        __syncthreads();
#pragma unroll
        for (int ks = 0; ks < 64; ks += 32) {
            const long ib = i0 + kb + ks;
            v16bf A = *(const v16bf*)(arow + ib + 16 * half);
#pragma unroll
            for (int c = 0; c < 8; ++c) {
                v16bf B = *(const v16bf*)(&T[(c * 16 + nn) * 64 + ks + 16 * half]);
                acc[c] = __builtin_amdgcn_wmma_f32_16x16x32_bf16(false, A, false, B,
                                                                 (short)0, acc[c], false, false);
            }
        }
    }
#pragma unroll
    for (int c = 0; c < 8; ++c)
#pragma unroll
        for (int j = 0; j < 8; ++j)
            atomicAdd(&mu_acc[(size_t)(16 * w + j + 8 * half) * HDIM + h0 + c * 16 + nn],
                      acc[c][j]);
}

// ------- final mu (unnormalized), stored transposed [H x K] bf16 -------
__global__ void __launch_bounds__(256)
k_mu_final(const float* __restrict__ mu_acc, const float* __restrict__ cr,
           bf16_t* __restrict__ muT)
{
    const int kk = (blockIdx.x * 256 + threadIdx.x) >> 5;
    const int lane = threadIdx.x & 31;
    if (kk >= KCLUS) return;
    const float inv = 1.0f / cr[kk];
#pragma unroll
    for (int c = 0; c < 16; ++c) {
        const int h = lane + 32 * c;
        muT[(size_t)h * KCLUS + kk] = (bf16_t)(mu_acc[(size_t)kk * HDIM + h] * inv);
    }
}

// ------- epilogue: s = r@mu, sigmoid, all four BCE dot-sums fused -------
__global__ void __launch_bounds__(256)
k_phase2(const bf16_t* __restrict__ rrm, const bf16_t* __restrict__ muT,
         const float* __restrict__ pos, const float* __restrict__ neg,
         const float* __restrict__ WsG, float* __restrict__ acc4)
{
    __shared__ float lws[HDIM];
    __shared__ float lacc[4];
    const int t = threadIdx.x, w = t >> 5, lane = t & 31;
    const int half = lane >> 4, nn = lane & 15;
    for (int i = t; i < HDIM; i += 256) lws[i] = WsG[i];
    if (t < 4) lacc[t] = 0.0f;
    __syncthreads();

    const long iw = (long)blockIdx.x * 128 + (long)w * 16;
    float pc[8], ncl[8], pg[8], ng[8];
#pragma unroll
    for (int j = 0; j < 8; ++j) { pc[j] = 0.f; ncl[j] = 0.f; pg[j] = 0.f; ng[j] = 0.f; }

    const bf16_t* arow = rrm + (iw + nn) * (long)KCLUS;

    for (int hb = 0; hb < HBLOCKS; ++hb) {
        v8f s[8];
#pragma unroll
        for (int c = 0; c < 8; ++c)
#pragma unroll
            for (int j = 0; j < 8; ++j) s[c][j] = 0.0f;
#pragma unroll
        for (int ks = 0; ks < KCLUS; ks += 32) {
            v16bf A = *(const v16bf*)(arow + ks + 16 * half);
#pragma unroll
            for (int c = 0; c < 8; ++c) {
                const int h = hb * 128 + c * 16 + nn;
                v16bf B = *(const v16bf*)(muT + (size_t)h * KCLUS + ks + 16 * half);
                s[c] = __builtin_amdgcn_wmma_f32_16x16x32_bf16(false, A, false, B,
                                                               (short)0, s[c], false, false);
            }
        }
#pragma unroll
        for (int c = 0; c < 8; ++c) {
            const int h = hb * 128 + c * 16 + nn;
            const float wsv = lws[h];
#pragma unroll
            for (int j = 0; j < 8; ++j) {
                const long node = iw + j + 8 * half;
                if (node < N_NODES) {
                    const float sig = sigmoidf_(s[c][j]);
                    const float pv = pos[node * (long)HDIM + h];
                    const float nv = neg[node * (long)HDIM + h];
                    pc[j] += pv * sig; ncl[j] += nv * sig;
                    pg[j] += pv * wsv; ng[j]  += nv * wsv;
                }
            }
        }
    }

    float t0 = 0.f, t1 = 0.f, t2 = 0.f, t3 = 0.f;
#pragma unroll
    for (int j = 0; j < 8; ++j) {
        float a = red16_add(pc[j]);
        float b = red16_add(ncl[j]);
        float c2 = red16_add(pg[j]);
        float d = red16_add(ng[j]);
        const long node = iw + j + 8 * half;
        if (nn == 0 && node < N_NODES) {
            t0 += softplusf(-c2);  // bce_ones(pos_graph)
            t1 += softplusf(d);    // bce_zeros(neg_graph)
            t2 += softplusf(-a);   // bce_ones(pos_cluster)
            t3 += softplusf(b);    // bce_zeros(neg_cluster)
        }
    }
    if (nn == 0) {
        atomicAdd(&lacc[0], t0); atomicAdd(&lacc[1], t1);
        atomicAdd(&lacc[2], t2); atomicAdd(&lacc[3], t3);
    }
    __syncthreads();
    if (t < 4) atomicAdd(&acc4[t], lacc[t]);
}

__global__ void k_finalize(const float* __restrict__ acc4, float* __restrict__ out)
{
    if (threadIdx.x == 0) {
        const float invn = 1.0f / (float)N_NODES;
        out[0] = 0.5f * (acc4[0] + acc4[1]) * invn + 0.5f * (acc4[2] + acc4[3]) * invn;
    }
}

// ---------------- host orchestration ----------------
extern "C" void kernel_launch(void* const* d_in, const int* in_sizes, int n_in,
                              void* d_out, int out_size, void* d_ws, size_t ws_size,
                              hipStream_t stream)
{
    (void)in_sizes; (void)n_in; (void)out_size; (void)ws_size;
    const float* positive = (const float*)d_in[0];
    const float* negative = (const float*)d_in[1];
    const float* initmu   = (const float*)d_in[2];
    const float* W        = (const float*)d_in[3];
    float* out = (float*)d_out;

    char* ws = (char*)d_ws;
    size_t off = 0;
    auto salloc = [&](size_t bytes) -> char* {
        char* p = ws + off;
        off = (off + bytes + 255) & ~(size_t)255;
        return p;
    };
    bf16_t* dn     = (bf16_t*)salloc((size_t)NPAD2 * HDIM * sizeof(bf16_t));
    bf16_t* rt     = (bf16_t*)salloc((size_t)KCLUS * NPAD2 * sizeof(bf16_t));
    bf16_t* rrm    = (bf16_t*)salloc((size_t)NPAD2 * KCLUS * sizeof(bf16_t));
    bf16_t* mun    = (bf16_t*)salloc((size_t)KCLUS * HDIM * sizeof(bf16_t));
    bf16_t* muT    = (bf16_t*)salloc((size_t)HDIM * KCLUS * sizeof(bf16_t));
    float*  mu_acc = (float*) salloc((size_t)KCLUS * HDIM * sizeof(float));
    float*  cr     = (float*) salloc(KCLUS * sizeof(float));
    float*  colsum = (float*) salloc(HDIM * sizeof(float));
    float*  Ws     = (float*) salloc(HDIM * sizeof(float));
    float*  acc4   = (float*) salloc(4 * sizeof(float));

    // per-call zero-init (graph-capture safe)
    hipMemsetAsync(dn + (size_t)N_NODES * HDIM, 0,
                   (size_t)(NPAD2 - N_NODES) * HDIM * sizeof(bf16_t), stream);
    hipMemsetAsync(rt, 0, (size_t)KCLUS * NPAD2 * sizeof(bf16_t), stream);
    hipMemsetAsync(rrm + (size_t)N_NODES * KCLUS, 0,
                   (size_t)(NPAD2 - N_NODES) * KCLUS * sizeof(bf16_t), stream);
    hipMemsetAsync(colsum, 0, HDIM * sizeof(float), stream);
    hipMemsetAsync(acc4, 0, 4 * sizeof(float), stream);

    k_normalize<<<(N_NODES + 7) / 8, 256, 0, stream>>>(positive, dn);
    k_colsum<<<(N_NODES + 255) / 256, 256, 0, stream>>>(positive, colsum);
    k_ws<<<HDIM / 8, 256, 0, stream>>>(W, colsum, Ws);

    for (int it = 0; it < 11; ++it) {
        k_mu_prep<<<KCLUS / 8, 256, 0, stream>>>(initmu, mu_acc, cr, mun, it == 0 ? 1 : 0);
        k_dist_softmax<<<NPAD / 256, 256, 0, stream>>>(dn, mun, rt, rrm, cr,
                                                       it == 10 ? 1 : 0);
        dim3 gm(NCHUNKS, HBLOCKS);
        k_mu_accum<<<gm, 256, 0, stream>>>(rt, dn, mu_acc);
    }

    k_mu_final<<<KCLUS / 8, 256, 0, stream>>>(mu_acc, cr, muT);
    k_phase2<<<NPAD / 128, 256, 0, stream>>>(rrm, muT, positive, negative, Ws, acc4);
    k_finalize<<<1, 32, 0, stream>>>(acc4, out);
}
